// HeLpER_76003741270026
// MI455X (gfx1250) — matrix operand order repaired and checked by reading, not compiled, not run on hardware
//
#include <hip/hip_runtime.h>
#include <hip/hip_bf16.h>
#include <math.h>

#define NUM_ENTS 100000
#define IN_DIM   200
#define OUT_DIM  2000
#define BATCH    512
#define OUT_CH   32
#define KER      9
#define L_IN     2000
#define L_OUT    1992
#define FC_LEN   63744      // 32*1992, multiple of 32
#define KPAD     224        // IN_DIM padded to multiple of 32
#define EPS      1e-5f

typedef __bf16 bf16_t;
typedef __bf16 v16bf __attribute__((ext_vector_type(16)));
typedef float  v8f   __attribute__((ext_vector_type(8)));
typedef unsigned int v4u __attribute__((ext_vector_type(4)));

union FragU { v4u q[2]; v16bf v; };

// A fragment (16x32 bf16): lane = row M (lane&15), half = lane>>4.
// element e: k = 16*half + e  -> two contiguous 16B loads.
__device__ inline v16bf load_a_frag(const bf16_t* rowk /* &A[m][kbase] */, int half) {
  FragU f;
  f.q[0] = *(const v4u*)(rowk + half * 16);
  f.q[1] = *(const v4u*)(rowk + half * 16 + 8);
  return f.v;
}

// B fragment (32x16 bf16): lane = column (lane&15), element e: k = 16*half + e.
// Source row-major contiguous in k; convert f32 -> bf16 in-register; zero-pad past krem.
__device__ inline v16bf make_b_rowmajor(const float* src, int krem) {
  v16bf r;
#pragma unroll
  for (int e = 0; e < 16; ++e) {
    float x = (e < krem) ? src[e] : 0.0f;
    r[e] = (bf16_t)x;
  }
  return r;
}

// Same, but source strided (column of a K-major matrix).
__device__ inline v16bf make_b_strided(const float* base, int stride, int krem) {
  v16bf r;
#pragma unroll
  for (int e = 0; e < 16; ++e) {
    float x = (e < krem) ? base[(long)e * stride] : 0.0f;
    r[e] = (bf16_t)x;
  }
  return r;
}

__device__ inline v8f wmma_bf16(v16bf a, v16bf b, v8f c) {
  return __builtin_amdgcn_wmma_f32_16x16x32_bf16(false, a, false, b, (short)0, c, false, false);
}

// ---- Stage 1: gather + bn0 + bf16 pack (512 x KPAD) -------------------------
__global__ void k_bn0(const float* __restrict__ E, const float* __restrict__ g,
                      const float* __restrict__ bt, const int* __restrict__ e1,
                      bf16_t* __restrict__ x0a) {
  int d = threadIdx.x;
  if (d >= KPAD) return;
  if (d < IN_DIM) {
    float s = 0.f, q = 0.f;
    for (int b = 0; b < BATCH; ++b) {
      float x = E[(long)e1[b] * IN_DIM + d];
      s += x; q += x * x;
    }
    float m  = s / BATCH;
    float v  = q / BATCH - m * m;
    float sc = rsqrtf(v + EPS) * g[d];
    float sh = bt[d] - m * sc;
    for (int b = 0; b < BATCH; ++b) {
      float x = E[(long)e1[b] * IN_DIM + d];
      x0a[b * KPAD + d] = (bf16_t)(x * sc + sh);
    }
  } else {
    for (int b = 0; b < BATCH; ++b) x0a[b * KPAD + d] = (bf16_t)0.0f;
  }
}

// ---- Stage 2: x1 = x0 @ e_W  (512x200 * 200x2000), WMMA bf16 ----------------
__global__ void k_gemm_x1(const bf16_t* __restrict__ A, const float* __restrict__ W,
                          float* __restrict__ X1) {
  int w = threadIdx.x >> 5, lane = threadIdx.x & 31;
  int gid = blockIdx.x * 8 + w;            // 4000 tiles = 32 mtiles x 125 ntiles
  int mtile = gid / 125, ntile = gid % 125;
  int half = lane >> 4, c = lane & 15;
  int m = mtile * 16 + c;
  v8f acc = {};
  for (int ks = 0; ks < KPAD / 32; ++ks) {
    int kk = ks * 32;
    v16bf a = load_a_frag(A + m * KPAD + kk, half);
    int kb = kk + half * 16;
    v16bf b = make_b_strided(W + (long)kb * OUT_DIM + (ntile * 16 + c), OUT_DIM, IN_DIM - kb);
    acc = wmma_bf16(a, b, acc);
  }
  int col = ntile * 16 + c;
#pragma unroll
  for (int r = 0; r < 8; ++r) {
    int row = mtile * 16 + r + 8 * half;
    X1[row * OUT_DIM + col] = acc[r];
  }
}

// ---- Stage 3: conv1d per-sample stats (sum, sumsq per channel) --------------
__global__ void k_conv_stats(const float* __restrict__ X1, const float* __restrict__ R,
                             const int* __restrict__ ridx, float* __restrict__ psum,
                             float* __restrict__ psq) {
  __shared__ float ls[256], lq[256];
  int b = blockIdx.x;
  int o = threadIdx.x & 31, chunk = threadIdx.x >> 5;
  const float* wv = R + (long)ridx[b] * OUT_CH * KER + o * KER;
  float wreg[KER];
#pragma unroll
  for (int k = 0; k < KER; ++k) wreg[k] = wv[k];
  const float* xr = X1 + (long)b * L_IN;
  float s = 0.f, q = 0.f;
  int l0 = chunk * (L_OUT / 8);
  for (int i = 0; i < L_OUT / 8; ++i) {
    int l = l0 + i;
    float acc = 0.f;
#pragma unroll
    for (int k = 0; k < KER; ++k) acc = fmaf(wreg[k], xr[l + k], acc);
    s += acc; q += acc * acc;
  }
  ls[threadIdx.x] = s; lq[threadIdx.x] = q;
  __syncthreads();
  for (int st = 128; st >= 32; st >>= 1) {
    if ((int)threadIdx.x < st) {
      ls[threadIdx.x] += ls[threadIdx.x + st];
      lq[threadIdx.x] += lq[threadIdx.x + st];
    }
    __syncthreads();
  }
  if (threadIdx.x < 32) {
    psum[b * 32 + threadIdx.x] = ls[threadIdx.x];
    psq [b * 32 + threadIdx.x] = lq[threadIdx.x];
  }
}

// ---- Stage 4: finalize bn1 (deterministic sequential reduction) -------------
__global__ void k_bn1_fin(const float* __restrict__ psum, const float* __restrict__ psq,
                          const float* __restrict__ g, const float* __restrict__ bt,
                          float* __restrict__ scale, float* __restrict__ shift) {
  int o = threadIdx.x;
  if (o >= OUT_CH) return;
  float s = 0.f, q = 0.f;
  for (int b = 0; b < BATCH; ++b) { s += psum[b * 32 + o]; q += psq[b * 32 + o]; }
  const float cnt = (float)BATCH * (float)L_OUT;
  float m = s / cnt, v = q / cnt - m * m;
  float sc = rsqrtf(v + EPS) * g[o];
  scale[o] = sc; shift[o] = bt[o] - m * sc;
}

// ---- Stage 5: conv1d again + bn1 normalize + bf16 pack ----------------------
__global__ void k_conv_norm(const float* __restrict__ X1, const float* __restrict__ R,
                            const int* __restrict__ ridx, const float* __restrict__ scale,
                            const float* __restrict__ shift, bf16_t* __restrict__ X2) {
  int b = blockIdx.x;
  int j = blockIdx.y * 256 + threadIdx.x;   // < 63744
  int o = j / L_OUT, l = j % L_OUT;
  const float* wv = R + (long)ridx[b] * OUT_CH * KER + o * KER;
  const float* xr = X1 + (long)b * L_IN;
  float acc = 0.f;
#pragma unroll
  for (int k = 0; k < KER; ++k) acc = fmaf(wv[k], xr[l + k], acc);
  X2[(long)b * FC_LEN + j] = (bf16_t)(acc * scale[o] + shift[o]);
}

// ---- Stage 6: x3 = x2 @ fc_W.T  (512x63744 * 63744x200), WMMA bf16 ----------
__global__ void k_gemm_x3(const bf16_t* __restrict__ A, const float* __restrict__ FCW,
                          float* __restrict__ X3) {
  int w = threadIdx.x >> 5, lane = threadIdx.x & 31;
  int gid = blockIdx.x * 8 + w;          // 104 waves = 8 mgroups x 13 ntiles
  int mg = gid / 13, ntile = gid % 13;
  int half = lane >> 4, c = lane & 15;
  int d = ntile * 16 + c;
  int dc = d < IN_DIM ? d : IN_DIM - 1;  // clamp; clamped columns never stored
  const float* bsrc0 = FCW + (long)dc * FC_LEN + half * 16;
  v8f acc[4] = {{}, {}, {}, {}};
  const bf16_t* arow[4];
#pragma unroll
  for (int mt = 0; mt < 4; ++mt)
    arow[mt] = A + (long)(mg * 64 + mt * 16 + c) * FC_LEN;
  for (int ks = 0; ks < FC_LEN / 32; ++ks) {
    int kk = ks * 32;
    v16bf b = make_b_rowmajor(bsrc0 + kk, 16);
    __builtin_prefetch(bsrc0 + kk + 64, 0, 1);   // global_prefetch_b8
#pragma unroll
    for (int mt = 0; mt < 4; ++mt) {
      v16bf a = load_a_frag(arow[mt] + kk, half);
      acc[mt] = wmma_bf16(a, b, acc[mt]);
    }
  }
  if (d < IN_DIM) {
#pragma unroll
    for (int mt = 0; mt < 4; ++mt)
#pragma unroll
      for (int r = 0; r < 8; ++r) {
        int row = mg * 64 + mt * 16 + r + 8 * half;
        X3[row * IN_DIM + d] = acc[mt][r];
      }
  }
}

// ---- Stage 7: bn2 + tanh + bf16 pack (512 x KPAD) ---------------------------
__global__ void k_bn2(const float* __restrict__ X3, const float* __restrict__ g,
                      const float* __restrict__ bt, bf16_t* __restrict__ X4) {
  int d = threadIdx.x;
  if (d >= KPAD) return;
  if (d < IN_DIM) {
    float s = 0.f, q = 0.f;
    for (int b = 0; b < BATCH; ++b) { float x = X3[b * IN_DIM + d]; s += x; q += x * x; }
    float m  = s / BATCH, v = q / BATCH - m * m;
    float sc = rsqrtf(v + EPS) * g[d], sh = bt[d] - m * sc;
    for (int b = 0; b < BATCH; ++b) {
      float x = X3[b * IN_DIM + d];
      X4[b * KPAD + d] = (bf16_t)tanhf(x * sc + sh);
    }
  } else {
    for (int b = 0; b < BATCH; ++b) X4[b * KPAD + d] = (bf16_t)0.0f;
  }
}

// ---- Stage 8: logits = x4 @ E.T + sigmoid  (512x200 * 200x100000) -----------
__global__ void k_gemm_logits(const bf16_t* __restrict__ A, const float* __restrict__ Emat,
                              float* __restrict__ out) {
  int w = threadIdx.x >> 5, lane = threadIdx.x & 31;
  int ntile = blockIdx.x * 8 + w;
  if (ntile >= NUM_ENTS / 16) return;
  int mg = blockIdx.y;                    // 8 groups of 64 rows
  int half = lane >> 4, c = lane & 15;
  int n = ntile * 16 + c;
  const float* bsrc = Emat + (long)n * IN_DIM;
  v8f acc[4] = {{}, {}, {}, {}};
  for (int ks = 0; ks < KPAD / 32; ++ks) {
    int kk = ks * 32;
    int kb = kk + half * 16;
    v16bf b = make_b_rowmajor(bsrc + kb, IN_DIM - kb);
#pragma unroll
    for (int mt = 0; mt < 4; ++mt) {
      v16bf a = load_a_frag(A + (long)(mg * 64 + mt * 16 + c) * KPAD + kk, half);
      acc[mt] = wmma_bf16(a, b, acc[mt]);
    }
  }
#pragma unroll
  for (int mt = 0; mt < 4; ++mt)
#pragma unroll
    for (int r = 0; r < 8; ++r) {
      int row = mg * 64 + mt * 16 + r + 8 * half;
      float z = acc[mt][r];
      out[(long)row * NUM_ENTS + ntile * 16 + c] = 1.0f / (1.0f + __expf(-z));
    }
}

extern "C" void kernel_launch(void* const* d_in, const int* in_sizes, int n_in,
                              void* d_out, int out_size, void* d_ws, size_t ws_size,
                              hipStream_t stream) {
  const float* E   = (const float*)d_in[0];
  const float* R   = (const float*)d_in[1];
  const float* eW  = (const float*)d_in[2];
  const float* fcW = (const float*)d_in[3];
  const float* g0  = (const float*)d_in[4];
  const float* b0  = (const float*)d_in[5];
  const float* g1  = (const float*)d_in[6];
  const float* b1  = (const float*)d_in[7];
  const float* g2  = (const float*)d_in[8];
  const float* b2  = (const float*)d_in[9];
  const int*   e1  = (const int*)d_in[10];
  const int*   ri  = (const int*)d_in[11];
  float* out = (float*)d_out;
  (void)in_sizes; (void)n_in; (void)out_size; (void)ws_size;

  char* ws = (char*)d_ws;
  size_t off = 0;
  auto alloc = [&](size_t bytes) -> char* {
    char* p = ws + off;
    off = (off + bytes + 255) & ~(size_t)255;
    return p;
  };
  bf16_t* x0a = (bf16_t*)alloc((size_t)BATCH * KPAD * 2);
  float*  x1  = (float*) alloc((size_t)BATCH * L_IN * 4);
  float*  psum= (float*) alloc((size_t)BATCH * OUT_CH * 4);
  float*  psq = (float*) alloc((size_t)BATCH * OUT_CH * 4);
  float*  sc1 = (float*) alloc(OUT_CH * 4);
  float*  sh1 = (float*) alloc(OUT_CH * 4);
  bf16_t* x2  = (bf16_t*)alloc((size_t)BATCH * FC_LEN * 2);
  float*  x3  = (float*) alloc((size_t)BATCH * IN_DIM * 4);
  bf16_t* x4  = (bf16_t*)alloc((size_t)BATCH * KPAD * 2);

  hipLaunchKernelGGL(k_bn0,        dim3(1),        dim3(256), 0, stream, E, g0, b0, e1, x0a);
  hipLaunchKernelGGL(k_gemm_x1,    dim3(500),      dim3(256), 0, stream, x0a, eW, x1);
  hipLaunchKernelGGL(k_conv_stats, dim3(512),      dim3(256), 0, stream, x1, R, ri, psum, psq);
  hipLaunchKernelGGL(k_bn1_fin,    dim3(1),        dim3(32),  0, stream, psum, psq, g1, b1, sc1, sh1);
  hipLaunchKernelGGL(k_conv_norm,  dim3(512, 249), dim3(256), 0, stream, x1, R, ri, sc1, sh1, x2);
  hipLaunchKernelGGL(k_gemm_x3,    dim3(13),       dim3(256), 0, stream, x2, fcW, x3);
  hipLaunchKernelGGL(k_bn2,        dim3(1),        dim3(256), 0, stream, x3, g2, b2, x4);
  hipLaunchKernelGGL(k_gemm_logits,dim3(782, 8),   dim3(256), 0, stream, x4, E, out);
}
